// Codebook_51110110822774
// MI455X (gfx1250) — matrix operand order repaired and checked by reading, not compiled, not run on hardware
//
#include <hip/hip_runtime.h>

typedef __attribute__((ext_vector_type(16))) _Float16 v16h;
typedef __attribute__((ext_vector_type(4)))  _Float16 v4h;
typedef __attribute__((ext_vector_type(8)))  float    v8f;

#define BB 16
#define TT 4096
#define DD 64
#define KK 1024
#define NN (BB * TT)            /* 65536 rows */
#define ZQ_SIZE (NN * DD)       /* 4194304 */
#define LOSS_OFF ZQ_SIZE
#define IDX_OFF (ZQ_SIZE + 1)
#define BETA 0.25f

#define WAVES 8
#define MT_PER_WAVE 4                       /* 4 M-tiles of 16 rows */
#define ROWS_PER_WAVE (MT_PER_WAVE * 16)    /* 64 */
#define ROWS_PER_BLOCK (WAVES * ROWS_PER_WAVE) /* 512 */
#define NBLOCKS (NN / ROWS_PER_BLOCK)       /* 128 */

__global__ void vq_init_kernel(float* out) {
    if (threadIdx.x == 0) out[LOSS_OFF] = 0.0f;
}

__global__ __launch_bounds__(256)
void vq_main_kernel(const float* __restrict__ z,
                    const float* __restrict__ emb,
                    float* __restrict__ out)
{
    // 128 KB f16 codebook + 4 KB norms + 2 KB idx scratch (<= 320 KB/WGP on CDNA5)
    __shared__ __align__(32) _Float16 embh[KK][DD];
    __shared__ float enorm[KK];
    __shared__ int   idx_sc[WAVES * ROWS_PER_WAVE];

    const int tid  = threadIdx.x;
    const int lane = tid & 31;
    const int wave = tid >> 5;

    // ---- stage codebook: f32 -> f16 into LDS, plus ||e||^2 in f32 ----
    for (int r = tid; r < KK; r += 256) {
        const float4* e4 = (const float4*)(emb + (size_t)r * DD);
        float ss = 0.0f;
        #pragma unroll
        for (int i = 0; i < 16; ++i) {
            float4 v = e4[i];
            ss += v.x * v.x + v.y * v.y + v.z * v.z + v.w * v.w;
            v4h h;
            h[0] = (_Float16)v.x; h[1] = (_Float16)v.y;
            h[2] = (_Float16)v.z; h[3] = (_Float16)v.w;
            *(v4h*)&embh[r][i * 4] = h;
        }
        enorm[r] = ss;
    }
    __syncthreads();

    // ---- load A tiles (4 x 16x64 f16, folded by -2) resident in VGPRs ----
    const int a_r  = lane & 15;            // A row within tile = lane % 16
    const int a_kb = (lane >> 4) * 16;     // K sub-block 0 / 16 per lane half
    const long rowbase = (long)blockIdx.x * ROWS_PER_BLOCK + (long)wave * ROWS_PER_WAVE;

    v16h A[MT_PER_WAVE][2];
    #pragma unroll
    for (int mt = 0; mt < MT_PER_WAVE; ++mt) {
        const long row = rowbase + mt * 16 + a_r;
        const float* zr = z + row * DD;
        #pragma unroll
        for (int ks = 0; ks < 2; ++ks) {
            const float* p = zr + ks * 32 + a_kb;
            float4 f0 = *(const float4*)(p);
            float4 f1 = *(const float4*)(p + 4);
            float4 f2 = *(const float4*)(p + 8);
            float4 f3 = *(const float4*)(p + 12);
            v16h a;
            a[0]  = (_Float16)(-2.0f * f0.x); a[1]  = (_Float16)(-2.0f * f0.y);
            a[2]  = (_Float16)(-2.0f * f0.z); a[3]  = (_Float16)(-2.0f * f0.w);
            a[4]  = (_Float16)(-2.0f * f1.x); a[5]  = (_Float16)(-2.0f * f1.y);
            a[6]  = (_Float16)(-2.0f * f1.z); a[7]  = (_Float16)(-2.0f * f1.w);
            a[8]  = (_Float16)(-2.0f * f2.x); a[9]  = (_Float16)(-2.0f * f2.y);
            a[10] = (_Float16)(-2.0f * f2.z); a[11] = (_Float16)(-2.0f * f2.w);
            a[12] = (_Float16)(-2.0f * f3.x); a[13] = (_Float16)(-2.0f * f3.y);
            a[14] = (_Float16)(-2.0f * f3.z); a[15] = (_Float16)(-2.0f * f3.w);
            A[mt][ks] = a;
        }
    }

    // ---- running min/argmin state ----
    float minv[MT_PER_WAVE][8];
    int   mini[MT_PER_WAVE][8];
    #pragma unroll
    for (int mt = 0; mt < MT_PER_WAVE; ++mt)
        #pragma unroll
        for (int v = 0; v < 8; ++v) { minv[mt][v] = 3.4e38f; mini[mt][v] = 0; }

    const int bcol = lane & 15;            // B column (code within tile)
    const int bkb  = (lane >> 4) * 16;     // K sub-block 0 / 16 per lane half

    // dist tile: acc = ||e||^2 + (-2z) . e  via two chained WMMAs, C init = ||e||^2
    auto compute_tile = [&](v8f (&acc)[MT_PER_WAVE], int ct) {
        const int c = ct * 16 + bcol;
        const _Float16* brow = &embh[c][0];
        v16h B0 = *(const v16h*)(brow + bkb);        // GEMM-K 0..31 slice
        v16h B1 = *(const v16h*)(brow + 32 + bkb);   // GEMM-K 32..63 slice
        const float en = enorm[c];
        v8f ci;
        #pragma unroll
        for (int v = 0; v < 8; ++v) ci[v] = en;
        #pragma unroll
        for (int mt = 0; mt < MT_PER_WAVE; ++mt) {
            v8f t = __builtin_amdgcn_wmma_f32_16x16x32_f16(false, A[mt][0], false, B0,
                                                           (short)0, ci, false, false);
            t = __builtin_amdgcn_wmma_f32_16x16x32_f16(false, A[mt][1], false, B1,
                                                       (short)0, t, false, false);
            acc[mt] = t;
        }
    };
    // min/argmin update from a previously computed tile (no WMMA hazard: results old)
    auto update_tile = [&](const v8f (&acc)[MT_PER_WAVE], int ct) {
        const int c = ct * 16 + bcol;
        #pragma unroll
        for (int mt = 0; mt < MT_PER_WAVE; ++mt)
            #pragma unroll
            for (int v = 0; v < 8; ++v) {
                float d = acc[mt][v];
                if (d < minv[mt][v]) { minv[mt][v] = d; mini[mt][v] = c; }
            }
    };

    // ---- software-pipelined scan over 64 code tiles (double-buffered accs) ----
    v8f accA[MT_PER_WAVE], accB[MT_PER_WAVE];
    compute_tile(accA, 0);
    for (int ct = 1; ct < 63; ct += 2) {
        compute_tile(accB, ct);         // WMMAs for tile ct ...
        update_tile(accA, ct - 1);      // ... overlap with cmps on tile ct-1
        compute_tile(accA, ct + 1);
        update_tile(accB, ct);
    }
    compute_tile(accB, 63);
    update_tile(accA, 62);
    update_tile(accB, 63);

    // ---- argmin across the 16 lanes of each half-wave (masks 1,2,4,8 stay in-half)
    #pragma unroll
    for (int mt = 0; mt < MT_PER_WAVE; ++mt) {
        #pragma unroll
        for (int v = 0; v < 8; ++v) {
            float mv = minv[mt][v];
            int   mi = mini[mt][v];
            #pragma unroll
            for (int s = 1; s <= 8; s <<= 1) {
                float ov = __shfl_xor(mv, s, 32);
                int   oi = __shfl_xor(mi, s, 32);
                if (ov < mv || (ov == mv && oi < mi)) { mv = ov; mi = oi; }
            }
            if ((lane & 15) == 0) {
                // lanes 0..15 own C rows 0..7; lanes 16..31 own rows 8..15
                int row_local = mt * 16 + (lane >> 4) * 8 + v;
                idx_sc[wave * ROWS_PER_WAVE + row_local] = mi;
            }
        }
    }
    __syncthreads();

    // ---- epilogue: gather f32 emb[idx] -> z_q, write idx, accumulate loss ----
    float lacc = 0.0f;
    #pragma unroll
    for (int rr = 0; rr < 2; ++rr) {
        const int row_local = lane * 2 + rr;
        const int myidx = idx_sc[wave * ROWS_PER_WAVE + row_local];
        const long grow = rowbase + row_local;
        out[IDX_OFF + grow] = (float)myidx;
        const float4* e4 = (const float4*)(emb + (size_t)myidx * DD);
        const float4* z4 = (const float4*)(z + grow * DD);
        float4* o4 = (float4*)(out + grow * DD);
        #pragma unroll
        for (int i = 0; i < 16; ++i) {
            float4 ev = e4[i];
            float4 zv = z4[i];
            o4[i] = ev;
            float dx = ev.x - zv.x, dy = ev.y - zv.y;
            float dz = ev.z - zv.z, dw = ev.w - zv.w;
            lacc += dx * dx + dy * dy + dz * dz + dw * dw;
        }
    }
    #pragma unroll
    for (int s = 16; s >= 1; s >>= 1) lacc += __shfl_xor(lacc, s, 32);
    if (lane == 0) {
        // loss = (beta+1)/(T*D) * sum_all (z_q - z)^2  (commit == codebook numerically)
        atomicAdd(out + LOSS_OFF, lacc * ((1.0f + BETA) / (float)(TT * DD)));
    }
}

extern "C" void kernel_launch(void* const* d_in, const int* in_sizes, int n_in,
                              void* d_out, int out_size, void* d_ws, size_t ws_size,
                              hipStream_t stream) {
    (void)in_sizes; (void)n_in; (void)out_size; (void)d_ws; (void)ws_size;
    const float* z   = (const float*)d_in[0];
    const float* emb = (const float*)d_in[1];
    float* out = (float*)d_out;
    vq_init_kernel<<<1, 64, 0, stream>>>(out);
    vq_main_kernel<<<NBLOCKS, 256, 0, stream>>>(z, emb, out);
}
